// PredatorAttnActor_13589276525224
// MI455X (gfx1250) — matrix operand order repaired and checked by reading, not compiled, not run on hardware
//
#include <hip/hip_runtime.h>

// ---------------------------------------------------------------------------
// Fully-fused predator-attention actor for MI455X (gfx1250, wave32, WMMA).
//
// ~44 GFLOP; fused design touches only ~8MB of HBM inputs + 256KB output ->
// compute bound on v_wmma_f32_16x16x32_bf16. One block per batch (512 blocks,
// 8 waves).  Flash-attention online softmax over 11 chunks of 128 entities
// (processed as 2x64-key sub-tiles to keep VGPRs < 256); all intermediates
// staged in LDS as bf16 (f32 accumulation).  All nine 64x64 weight matrices
// live in LDS in pre-packed WMMA-B fragment order, so the steady-state inner
// loop is WMMA + DS only.
// ---------------------------------------------------------------------------

#define DEV __device__ __forceinline__

#ifndef __has_builtin
#define __has_builtin(x) 0
#endif

// Try the native packed f32->bf16 conversion instruction (gfx12-family VALU).
// Set to 0 if the assembler rejects the mnemonic.
#define USE_CVT_PK_BF16_ASM 1

typedef __attribute__((ext_vector_type(16))) __bf16 bf16x16;
typedef __attribute__((ext_vector_type(8)))  float  f32x8;

union FragU { unsigned int w[8]; bf16x16 v; };

#if USE_CVT_PK_BF16_ASM
DEV unsigned int pack2(float lo, float hi) {
  unsigned int r;
  asm("v_cvt_pk_bf16_f32 %0, %1, %2" : "=v"(r) : "v"(lo), "v"(hi));
  return r;
}
#else
DEV unsigned int f2bf_(float f) {
  unsigned int u = __builtin_bit_cast(unsigned int, f);
  return (u + 0x7FFFu + ((u >> 16) & 1u)) >> 16;
}
DEV unsigned int pack2(float lo, float hi) { return f2bf_(lo) | (f2bf_(hi) << 16); }
#endif

#if __has_builtin(__builtin_amdgcn_exp2f)
DEV float fexp2(float x) { return __builtin_amdgcn_exp2f(x); }
#else
DEV float fexp2(float x) { return exp2f(x); }
#endif

DEV f32x8 wmma_bf16(bf16x16 a, bf16x16 b, f32x8 c) {
  // D = A(16x32) * B(32x16) + C, f32 accumulate
  return __builtin_amdgcn_wmma_f32_16x16x32_bf16(false, a, false, b, (short)0, c,
                                                 false, false);
}

// A fragment (16x32, MxK) from row-major bf16 LDS tile (stride in 32-bit words).
// Contiguous-k layout (matches B): lane L: m=L&15, half=L>>4; VGPR j: K = 16*half + 2*j.
DEV bf16x16 load_fragA(const unsigned int* lds, int strideW, int rowbase, int kcol0,
                       int lane) {
  const int m = lane & 15, half = lane >> 4;
  const unsigned int* row = lds + (rowbase + m) * strideW;
  FragU f;
#pragma unroll
  for (int j = 0; j < 8; ++j) {
    int k = kcol0 + 16 * half + 2 * j;
    f.w[j] = row[k >> 1];
  }
  return f.v;
}

// B fragment (32x16, KxN) where LDS stores element B[K][n] at row n, col K.
// Lane L: n=L&15, half=L>>4; VGPR j: K = kcol0 + 16*half + 2*j.
DEV bf16x16 load_fragB(const unsigned int* lds, int strideW, int nrowbase, int kcol0,
                       int lane) {
  const int n = lane & 15, half = lane >> 4;
  const unsigned int* p = lds + (nrowbase + n) * strideW + (kcol0 >> 1) + 8 * half;
  FragU f;
#pragma unroll
  for (int j = 0; j < 8; ++j) f.w[j] = p[j];
  return f.v;
}

// Pre-packed weight B fragment (LDS resident; lane-consecutive words).
DEV bf16x16 load_fragW(const unsigned int* wl, int kt, int nt, int lane) {
  const unsigned int* base = wl + (kt * 4 + nt) * 256 + lane;
  FragU f;
#pragma unroll
  for (int j = 0; j < 8; ++j) f.w[j] = base[j * 32];
  return f.v;
}

// C/D fragment (16x16 f32) -> row-major bf16 LDS (stride in bf16 elems).
// lane L: n=L&15, half=L>>4; VGPR r -> row m = r + 8*half.
DEV void storeC(unsigned short* lds, int strideH, int rowbase, int colbase,
                const f32x8& c, int lane) {
  const int n = lane & 15, half = lane >> 4;
  unsigned short* p = lds + (rowbase + 8 * half) * strideH + colbase + n;
#pragma unroll
  for (int r = 0; r < 8; r += 2) {
    unsigned int w = pack2(c[r], c[r + 1]);
    p[(r + 0) * strideH] = (unsigned short)(w & 0xffffu);
    p[(r + 1) * strideH] = (unsigned short)(w >> 16);
  }
}
// transposed store: element (row m=key, col n=d) -> lds[d][key]
DEV void storeC_T(unsigned short* lds, int strideH, int drowbase, int keycolbase,
                  const f32x8& c, int lane) {
  const int n = lane & 15, half = lane >> 4;
  unsigned short* p = lds + (drowbase + n) * strideH + keycolbase + 8 * half;
#pragma unroll
  for (int r = 0; r < 8; r += 2) {
    unsigned int w = pack2(c[r], c[r + 1]);
    p[r + 0] = (unsigned short)(w & 0xffffu);
    p[r + 1] = (unsigned short)(w >> 16);
  }
}

DEV float rmax16(float v) {
#pragma unroll
  for (int m = 1; m < 16; m <<= 1) v = fmaxf(v, __shfl_xor(v, m, 32));
  return v;
}
DEV float rsum16(float v) {
#pragma unroll
  for (int m = 1; m < 16; m <<= 1) v += __shfl_xor(v, m, 32);
  return v;
}

// ---------------------------------------------------------------------------
// Kernel 0: pack nine 64x64 f32 weight matrices into WMMA-B-fragment word
// order (bf16 pairs).  dst[mat*2048 + tile*256 + j*32 + lane].
// ---------------------------------------------------------------------------
__global__ __launch_bounds__(256) void pack_weights(
    const float* __restrict__ m0, const float* __restrict__ m1,
    const float* __restrict__ m2, const float* __restrict__ m3,
    const float* __restrict__ m4, const float* __restrict__ m5,
    const float* __restrict__ m6, const float* __restrict__ m7,
    const float* __restrict__ m8, unsigned int* __restrict__ dst) {
  const float* src;
  switch (blockIdx.x) {
    case 0: src = m0; break; case 1: src = m1; break; case 2: src = m2; break;
    case 3: src = m3; break; case 4: src = m4; break; case 5: src = m5; break;
    case 6: src = m6; break; case 7: src = m7; break; default: src = m8; break;
  }
  unsigned int* o = dst + (size_t)blockIdx.x * 2048;
  for (int i = threadIdx.x; i < 2048; i += 256) {
    int tile = i >> 8;            // kt*4 + nt
    int j = (i >> 5) & 7;
    int ln = i & 31;
    int kt = tile >> 2, nt = tile & 3;
    int K = kt * 32 + 16 * (ln >> 4) + 2 * j;
    int n = nt * 16 + (ln & 15);
    o[i] = pack2(src[K * 64 + n], src[(K + 1) * 64 + n]);
  }
}

// ---------------------------------------------------------------------------
// Main fused kernel: one block (256 thr = 8 waves) per batch element.
// ---------------------------------------------------------------------------
__global__ __launch_bounds__(256) void predattn_fused(
    const float* __restrict__ pred, const float* __restrict__ prey,
    const float* __restrict__ obst, const float* __restrict__ emb,
    const float* __restrict__ e0w1, const float* __restrict__ e0b1,
    const float* __restrict__ e0b2, const float* __restrict__ e1w1,
    const float* __restrict__ e1b1, const float* __restrict__ e1b2,
    const float* __restrict__ e2w1, const float* __restrict__ e2b1,
    const float* __restrict__ e2b2, const float* __restrict__ bq,
    const float* __restrict__ bk, const float* __restrict__ bv,
    const float* __restrict__ wp, const float* __restrict__ bp,
    const float* __restrict__ bo, const float* __restrict__ nb1,
    const float* __restrict__ nb2, const float* __restrict__ nw3,
    const float* __restrict__ nb3, const unsigned int* __restrict__ wpk,
    float* __restrict__ out) {
  constexpr int NPc = 128, NYc = 1024, NOc = 256;
  constexpr float SCALE = 0.125f * 1.4426950408889634f;  // 1/sqrt(D) * log2(e)

  extern __shared__ unsigned char smem[];
  // PX: X / Q / P staging, [128 rows][66 bf16] (stride 33 words)
  unsigned int* PX = reinterpret_cast<unsigned int*>(smem);   // 128*33 words
  unsigned int* KL = PX + 128 * 33;                           // 128*33 words
  unsigned int* VL = KL + 128 * 33;                           // 64*65 words
  unsigned int* WL = VL + 64 * 65;                            // 9*2048 words
  float* POS = reinterpret_cast<float*>(WL + 9 * 2048);       // 128*2 f32
  unsigned short* PXh = reinterpret_cast<unsigned short*>(PX);
  unsigned short* KLh = reinterpret_cast<unsigned short*>(KL);
  unsigned short* VLh = reinterpret_cast<unsigned short*>(VL);

  const int b = blockIdx.x;
  const int lane = threadIdx.x & 31;
  const int wave = threadIdx.x >> 5;
  const int nlo = lane & 15, half = lane >> 4;
  const int erow0 = wave * 16;  // this wave's 16 rows (queries / chunk entities)

  // ---- copy pre-packed weights into LDS once (served by ds_load afterwards)
  for (int i = threadIdx.x; i < 9 * 2048; i += 256) WL[i] = wpk[i];
  __syncthreads();

  const unsigned int* W_e0w2 = WL + 0 * 2048;
  const unsigned int* W_e1w2 = WL + 1 * 2048;
  const unsigned int* W_e2w2 = WL + 2 * 2048;
  const unsigned int* W_wq   = WL + 3 * 2048;
  const unsigned int* W_wk   = WL + 4 * 2048;
  const unsigned int* W_wv   = WL + 5 * 2048;
  const unsigned int* W_wo   = WL + 6 * 2048;
  const unsigned int* W_nw1  = WL + 7 * 2048;
  const unsigned int* W_nw2  = WL + 8 * 2048;

  // ---------------- pre-phase: q, qp, qself for this wave's 16 queries ------
  float qp0r[8], qp1r[8], qsr[8];
  bf16x16 aq0, aq1;
  {
    const float* sp = pred + (size_t)b * NPc * 2;
    bf16x16 ha0, ha1;
    {
      float s0 = sp[(erow0 + nlo) * 2 + 0];
      float s1 = sp[(erow0 + nlo) * 2 + 1];
#pragma unroll
      for (int kt = 0; kt < 2; ++kt) {
        FragU f;
#pragma unroll
        for (int j = 0; j < 8; ++j) {
          int k = kt * 32 + 16 * half + 2 * j;
          float h0 = fmaxf(e0b1[k] + s0 * e0w1[k] + s1 * e0w1[64 + k], 0.f);
          float h1 = fmaxf(e0b1[k + 1] + s0 * e0w1[k + 1] + s1 * e0w1[64 + k + 1], 0.f);
          f.w[j] = pack2(h0, h1);
        }
        if (kt == 0) ha0 = f.v; else ha1 = f.v;
      }
    }
#pragma unroll
    for (int nt = 0; nt < 4; ++nt) {
      f32x8 acc = {};
      acc = wmma_bf16(ha0, load_fragW(W_e0w2, 0, nt, lane), acc);
      acc = wmma_bf16(ha1, load_fragW(W_e0w2, 1, nt, lane), acc);
      float bias = e0b2[nt * 16 + nlo] + emb[nt * 16 + nlo];
#pragma unroll
      for (int r = 0; r < 8; ++r) acc[r] += bias;
      storeC(PXh, 66, erow0, nt * 16, acc, lane);
    }
    __syncthreads();
    bf16x16 xa0 = load_fragA(PX, 33, erow0, 0, lane);
    bf16x16 xa1 = load_fragA(PX, 33, erow0, 32, lane);
    __syncthreads();
    f32x8 qt[4];
#pragma unroll
    for (int nt = 0; nt < 4; ++nt) {
      f32x8 acc = {};
      acc = wmma_bf16(xa0, load_fragW(W_wq, 0, nt, lane), acc);
      acc = wmma_bf16(xa1, load_fragW(W_wq, 1, nt, lane), acc);
      float bias = bq[nt * 16 + nlo];
#pragma unroll
      for (int r = 0; r < 8; ++r) acc[r] += bias;
      qt[nt] = acc;
    }
    // qp = q @ wp.T (2 comps), qbp = q . bp, qself = qp . pos_q + qbp
#pragma unroll
    for (int r = 0; r < 8; ++r) {
      float p0 = 0.f, p1 = 0.f, pb = 0.f;
#pragma unroll
      for (int nt = 0; nt < 4; ++nt) {
        int d = nt * 16 + nlo;
        p0 += qt[nt][r] * wp[d];
        p1 += qt[nt][r] * wp[64 + d];
        pb += qt[nt][r] * bp[d];
      }
      p0 = rsum16(p0); p1 = rsum16(p1); pb = rsum16(pb);
      int qrow = erow0 + r + 8 * half;
      float s0 = sp[qrow * 2 + 0], s1 = sp[qrow * 2 + 1];
      qp0r[r] = p0; qp1r[r] = p1;
      qsr[r] = p0 * s0 + p1 * s1 + pb;
    }
    // stage q as bf16 -> persistent A fragments
#pragma unroll
    for (int nt = 0; nt < 4; ++nt) storeC(PXh, 66, erow0, nt * 16, qt[nt], lane);
    __syncthreads();
    aq0 = load_fragA(PX, 33, erow0, 0, lane);
    aq1 = load_fragA(PX, 33, erow0, 32, lane);
    __syncthreads();
  }

  // ---------------- flash-attention state ----------------
  f32x8 o0 = {}, o1 = {}, o2 = {}, o3 = {};
  float mr[8], lr[8];
#pragma unroll
  for (int r = 0; r < 8; ++r) { mr[r] = -1e30f; lr[r] = 0.f; }

  // ---------------- chunk loop: 11 chunks of 128 entities -------------------
  for (int c = 0; c < 11; ++c) {
    const float* sptr; int comps;
    const float *w1, *b1, *b2, *embt; const unsigned int* w2p;
    if (c == 0) {
      sptr = pred + (size_t)b * NPc * 2; comps = 2;
      w1 = e0w1; b1 = e0b1; b2 = e0b2; embt = emb; w2p = W_e0w2;
    } else if (c <= 8) {
      sptr = prey + (size_t)b * NYc * 2 + (size_t)(c - 1) * 128 * 2; comps = 2;
      w1 = e1w1; b1 = e1b1; b2 = e1b2; embt = emb + 64; w2p = W_e1w2;
    } else {
      sptr = obst + (size_t)b * NOc * 3 + (size_t)(c - 9) * 128 * 3; comps = 3;
      w1 = e2w1; b1 = e2b1; b2 = e2b2; embt = emb + 128; w2p = W_e2w2;
    }

    // prefetch next chunk's entity states into the cache hierarchy
    if (c + 1 < 11 && threadIdx.x < 12) {
      const char* np;
      if (c + 1 <= 8)
        np = (const char*)(prey + (size_t)b * NYc * 2 + (size_t)c * 128 * 2);
      else
        np = (const char*)(obst + (size_t)b * NOc * 3 + (size_t)(c + 1 - 9) * 128 * 3);
      __builtin_prefetch(np + threadIdx.x * 128, 0, 0);
    }

    // stage positions (first 2 comps) for the chunk
    {
      int i = threadIdx.x >> 1, comp = threadIdx.x & 1;
      POS[threadIdx.x] = sptr[i * comps + comp];
    }

    // ---- encoder layer 1 (VALU, directly in A-fragment layout) ----
    bf16x16 ha0, ha1;
    {
      float s0 = sptr[(erow0 + nlo) * comps + 0];
      float s1 = sptr[(erow0 + nlo) * comps + 1];
      float s2 = (comps == 3) ? sptr[(erow0 + nlo) * comps + 2] : 0.f;
#pragma unroll
      for (int kt = 0; kt < 2; ++kt) {
        FragU f;
#pragma unroll
        for (int j = 0; j < 8; ++j) {
          int k = kt * 32 + 16 * half + 2 * j;
          float h0 = b1[k] + s0 * w1[k] + s1 * w1[64 + k];
          float h1 = b1[k + 1] + s0 * w1[k + 1] + s1 * w1[64 + k + 1];
          if (comps == 3) { h0 += s2 * w1[128 + k]; h1 += s2 * w1[128 + k + 1]; }
          f.w[j] = pack2(fmaxf(h0, 0.f), fmaxf(h1, 0.f));
        }
        if (kt == 0) ha0 = f.v; else ha1 = f.v;
      }
    }
    // ---- x = h @ w2 + b2 + emb[type]  -> stage in X ----
#pragma unroll
    for (int nt = 0; nt < 4; ++nt) {
      f32x8 acc = {};
      acc = wmma_bf16(ha0, load_fragW(w2p, 0, nt, lane), acc);
      acc = wmma_bf16(ha1, load_fragW(w2p, 1, nt, lane), acc);
      float bias = b2[nt * 16 + nlo] + embt[nt * 16 + nlo];
#pragma unroll
      for (int r = 0; r < 8; ++r) acc[r] += bias;
      storeC(PXh, 66, erow0, nt * 16, acc, lane);
    }
    __syncthreads();
    bf16x16 xa0 = load_fragA(PX, 33, erow0, 0, lane);
    bf16x16 xa1 = load_fragA(PX, 33, erow0, 32, lane);
    // ---- k = x@wk + bk -> KL (row-major); v = x@wv + bv -> VL (transposed) --
#pragma unroll
    for (int nt = 0; nt < 4; ++nt) {
      f32x8 ka = {};
      ka = wmma_bf16(xa0, load_fragW(W_wk, 0, nt, lane), ka);
      ka = wmma_bf16(xa1, load_fragW(W_wk, 1, nt, lane), ka);
      f32x8 va = {};
      va = wmma_bf16(xa0, load_fragW(W_wv, 0, nt, lane), va);
      va = wmma_bf16(xa1, load_fragW(W_wv, 1, nt, lane), va);
      float kb_ = bk[nt * 16 + nlo], vb_ = bv[nt * 16 + nlo];
#pragma unroll
      for (int r = 0; r < 8; ++r) { ka[r] += kb_; va[r] += vb_; }
      storeC(KLh, 66, erow0, nt * 16, ka, lane);
      storeC_T(VLh, 130, nt * 16, erow0, va, lane);
    }
    __syncthreads();

    // ---- attention in two 64-key halves (keeps live scores at 4 tiles) ----
#pragma unroll
    for (int hh = 0; hh < 2; ++hh) {
      const int kb = hh * 64;
      f32x8 S[4];
#pragma unroll
      for (int t = 0; t < 4; ++t) {
        f32x8 acc = {};
        acc = wmma_bf16(aq0, load_fragB(KL, 33, kb + t * 16, 0, lane), acc);
        acc = wmma_bf16(aq1, load_fragB(KL, 33, kb + t * 16, 32, lane), acc);
        float px0 = POS[(kb + t * 16 + nlo) * 2 + 0];
        float px1 = POS[(kb + t * 16 + nlo) * 2 + 1];
#pragma unroll
        for (int r = 0; r < 8; ++r)
          acc[r] = (acc[r] + qsr[r] - qp0r[r] * px0 - qp1r[r] * px1) * SCALE;
        S[t] = acc;
      }
      // online softmax (base-2 domain)
#pragma unroll
      for (int r = 0; r < 8; ++r) {
        float mx = fmaxf(fmaxf(S[0][r], S[1][r]), fmaxf(S[2][r], S[3][r]));
        mx = rmax16(mx);
        float mnew = fmaxf(mr[r], mx);
        float corr = fexp2(mr[r] - mnew);
        mr[r] = mnew;
        float rs = 0.f;
#pragma unroll
        for (int t = 0; t < 4; ++t) {
          float p = fexp2(S[t][r] - mnew);
          S[t][r] = p;
          rs += p;
        }
        rs = rsum16(rs);
        lr[r] = lr[r] * corr + rs;
        o0[r] *= corr; o1[r] *= corr; o2[r] *= corr; o3[r] *= corr;
      }
      // P (16x64) -> bf16 LDS (reuses X region), then o += P @ V
#pragma unroll
      for (int t = 0; t < 4; ++t) storeC(PXh, 66, erow0, t * 16, S[t], lane);
      __syncthreads();
#pragma unroll
      for (int kt = 0; kt < 2; ++kt) {
        bf16x16 pa = load_fragA(PX, 33, erow0, kt * 32, lane);
        o0 = wmma_bf16(pa, load_fragB(VL, 65, 0, kb + kt * 32, lane), o0);
        o1 = wmma_bf16(pa, load_fragB(VL, 65, 16, kb + kt * 32, lane), o1);
        o2 = wmma_bf16(pa, load_fragB(VL, 65, 32, kb + kt * 32, lane), o2);
        o3 = wmma_bf16(pa, load_fragB(VL, 65, 48, kb + kt * 32, lane), o3);
      }
      __syncthreads();
    }
  }

  // ---------------- epilogue: normalize, @wo, FFN, tanh ---------------------
#pragma unroll
  for (int r = 0; r < 8; ++r) {
    float inv = 1.0f / lr[r];
    o0[r] *= inv; o1[r] *= inv; o2[r] *= inv; o3[r] *= inv;
  }
  storeC(PXh, 66, erow0, 0, o0, lane);
  storeC(PXh, 66, erow0, 16, o1, lane);
  storeC(PXh, 66, erow0, 32, o2, lane);
  storeC(PXh, 66, erow0, 48, o3, lane);
  __syncthreads();
  {  // out = o @ wo + bo
    bf16x16 a0 = load_fragA(PX, 33, erow0, 0, lane);
    bf16x16 a1 = load_fragA(PX, 33, erow0, 32, lane);
    __syncthreads();
#pragma unroll
    for (int nt = 0; nt < 4; ++nt) {
      f32x8 acc = {};
      acc = wmma_bf16(a0, load_fragW(W_wo, 0, nt, lane), acc);
      acc = wmma_bf16(a1, load_fragW(W_wo, 1, nt, lane), acc);
      float bias = bo[nt * 16 + nlo];
#pragma unroll
      for (int r = 0; r < 8; ++r) acc[r] += bias;
      storeC(PXh, 66, erow0, nt * 16, acc, lane);
    }
  }
  __syncthreads();
  {  // h1 = relu(out @ n_w1 + n_b1)
    bf16x16 a0 = load_fragA(PX, 33, erow0, 0, lane);
    bf16x16 a1 = load_fragA(PX, 33, erow0, 32, lane);
    __syncthreads();
#pragma unroll
    for (int nt = 0; nt < 4; ++nt) {
      f32x8 acc = {};
      acc = wmma_bf16(a0, load_fragW(W_nw1, 0, nt, lane), acc);
      acc = wmma_bf16(a1, load_fragW(W_nw1, 1, nt, lane), acc);
      float bias = nb1[nt * 16 + nlo];
#pragma unroll
      for (int r = 0; r < 8; ++r) acc[r] = fmaxf(acc[r] + bias, 0.f);
      storeC(PXh, 66, erow0, nt * 16, acc, lane);
    }
  }
  __syncthreads();
  {  // h2 = relu(h1 @ n_w2 + n_b2); y = tanh(h2 . n_w3 + n_b3)
    bf16x16 a0 = load_fragA(PX, 33, erow0, 0, lane);
    bf16x16 a1 = load_fragA(PX, 33, erow0, 32, lane);
    float w3v[4];
#pragma unroll
    for (int nt = 0; nt < 4; ++nt) w3v[nt] = nw3[nt * 16 + nlo];
    float ysum[8];
#pragma unroll
    for (int r = 0; r < 8; ++r) ysum[r] = 0.f;
#pragma unroll
    for (int nt = 0; nt < 4; ++nt) {
      f32x8 acc = {};
      acc = wmma_bf16(a0, load_fragW(W_nw2, 0, nt, lane), acc);
      acc = wmma_bf16(a1, load_fragW(W_nw2, 1, nt, lane), acc);
      float bias = nb2[nt * 16 + nlo];
#pragma unroll
      for (int r = 0; r < 8; ++r)
        ysum[r] += fmaxf(acc[r] + bias, 0.f) * w3v[nt];
    }
    float b3 = nb3[0];
#pragma unroll
    for (int r = 0; r < 8; ++r) {
      float s = rsum16(ysum[r]);
      if (nlo == 0)
        out[(size_t)b * 128 + erow0 + r + 8 * half] = tanhf(s + b3);
    }
  }
}

// ---------------------------------------------------------------------------
extern "C" void kernel_launch(void* const* d_in, const int* in_sizes, int n_in,
                              void* d_out, int out_size, void* d_ws, size_t ws_size,
                              hipStream_t stream) {
  const float* pred = (const float*)d_in[0];
  const float* prey = (const float*)d_in[1];
  const float* obst = (const float*)d_in[2];
  // d_in[3] = prey_is_alive: unused (reference mask is identically false)
  const float* emb  = (const float*)d_in[4];
  const float* e0w1 = (const float*)d_in[5];
  const float* e0b1 = (const float*)d_in[6];
  const float* e0w2 = (const float*)d_in[7];
  const float* e0b2 = (const float*)d_in[8];
  const float* e1w1 = (const float*)d_in[9];
  const float* e1b1 = (const float*)d_in[10];
  const float* e1w2 = (const float*)d_in[11];
  const float* e1b2 = (const float*)d_in[12];
  const float* e2w1 = (const float*)d_in[13];
  const float* e2b1 = (const float*)d_in[14];
  const float* e2w2 = (const float*)d_in[15];
  const float* e2b2 = (const float*)d_in[16];
  const float* wq = (const float*)d_in[17];
  const float* bq = (const float*)d_in[18];
  const float* wk = (const float*)d_in[19];
  const float* bk = (const float*)d_in[20];
  const float* wv = (const float*)d_in[21];
  const float* bv = (const float*)d_in[22];
  const float* wp = (const float*)d_in[23];
  const float* bp = (const float*)d_in[24];
  const float* wo = (const float*)d_in[25];
  const float* bo = (const float*)d_in[26];
  const float* nw1 = (const float*)d_in[27];
  const float* nb1 = (const float*)d_in[28];
  const float* nw2 = (const float*)d_in[29];
  const float* nb2 = (const float*)d_in[30];
  const float* nw3 = (const float*)d_in[31];
  const float* nb3 = (const float*)d_in[32];
  (void)in_sizes; (void)n_in; (void)out_size; (void)ws_size;

  unsigned int* wpk = (unsigned int*)d_ws;  // 9 * 2048 words = 72 KB

  pack_weights<<<9, 256, 0, stream>>>(e0w2, e1w2, e2w2, wq, wk, wv, wo, nw1, nw2,
                                      wpk);

  const size_t smem =
      (size_t)(128 * 33 + 128 * 33 + 64 * 65 + 9 * 2048) * 4 +
      128 * 2 * sizeof(float);
  predattn_fused<<<512, 256, smem, stream>>>(
      pred, prey, obst, emb, e0w1, e0b1, e0b2, e1w1, e1b1, e1b2, e2w1, e2b1, e2b2,
      bq, bk, bv, wp, bp, bo, nb1, nb2, nw3, nb3, wpk, (float*)d_out);
}
